// MemoryModule_9663676416167
// MI455X (gfx1250) — compile-verified, matrix operand order checked by reading, not run on hardware
//
#include <hip/hip_runtime.h>
#include <hip/hip_bf16.h>

// ---------- problem constants ----------
#define CDIM   512      // feature dim (K of pass 1, N of pass 3)
#define MDIM   2048     // memory rows (N of pass 1, K of pass 3)
#define SWEEP  256      // M columns processed per sweep
#define NSWEEP 8        // MDIM / SWEEP
#define NTOK   4096     // tokens per batch (64*64)
#define RROWS  64       // token rows per workgroup (4 row-tiles of 16)

typedef __attribute__((ext_vector_type(16))) _Float16 v16h;
typedef __attribute__((ext_vector_type(8)))  float    v8f;

// 16-element f16 WMMA operand from pre-swizzled global tiles:
// lane base p, two contiguous 8-elem chunks at p and p+8.
__device__ __forceinline__ v16h ld_sw32(const _Float16* p) {
    union { uint4 q[2]; v16h v; } u;
    u.q[0] = *(const uint4*)(p);
    u.q[1] = *(const uint4*)(p + 8);
    return u.v;
}
// 16-element f16 operand from a row-major LDS buffer, honoring the CDNA5
// 16-bit K interleave: caller passes p = row + k + lh*8; chunks at p, p+16.
__device__ __forceinline__ v16h ld_row32(const _Float16* p) {
    union { uint4 q[2]; v16h v; } u;
    u.q[0] = *(const uint4*)(p);
    u.q[1] = *(const uint4*)(p + 16);
    return u.v;
}

// ---------- prep: convert memory bank to f16, pre-swizzled WMMA B-tiles ----
// Swizzled tile = 512 f16: lane l owns elements [l*16, l*16+16) in register
// order (lanes 0-15 hold K {0-7,16-23}, lanes 16-31 hold K {8-15,24-31}).
__global__ __launch_bounds__(256)
void mem_swizzle_f16_kernel(const float* __restrict__ mem,
                            _Float16* __restrict__ mhW,   // pass-1 B (K = C)
                            _Float16* __restrict__ mtW)   // pass-3 B (K = M)
{
    const int o    = blockIdx.x * 256 + threadIdx.x;   // 0 .. 2048*512-1
    const int e    = o & 511;
    const int tile = o >> 9;
    const int l    = e >> 4;
    const int i    = e & 15;
    const int ll   = l & 15;
    const int lh   = l >> 4;
    const int ko   = lh * 8 + (i < 8 ? i : i + 8);     // K interleave

    {   // pass-1 B operand: column n = memory row m, K = channel c
        const int tm = tile >> 4, tk = tile & 15;       // 128 m-tiles x 16 k-tiles
        mhW[o] = (_Float16)mem[(tm * 16 + ll) * CDIM + tk * 32 + ko];
    }
    {   // pass-3 B operand: column n = channel c, K = memory row m
        const int tc = tile >> 6, tk2 = tile & 63;      // 32 c-tiles x 64 k-tiles
        mtW[o] = (_Float16)mem[(tk2 * 32 + ko) * CDIM + tc * 16 + ll];
    }
}

// ---------- LDS layout ----------
#define SS_STRIDE 264                        // padded f32 row stride (64 rows)
#define SA_STRIDE 520                        // padded f16 row stride (64 rows)
#define OFF_A     67584                      // 64*264*4
#define OFF_STAT  134144                     // OFF_A + 64*520*2
#define OFF_PART  135168                     // OFF_STAT + 1024
#define SMEM_BYTES 136192                    // OFF_PART + 1024
// sO staging (512 x 65 f32 = 133,120 B) reuses [0, OFF_STAT) after sweeps.

__global__ __launch_bounds__(256)
void mem_attn_kernel(const float* __restrict__ F,
                     const _Float16* __restrict__ MhS,
                     const _Float16* __restrict__ MtS,
                     float* __restrict__ Out)
{
    __shared__ __align__(16) char smem[SMEM_BYTES];
    float*    sS    = (float*)smem;                    // 64 x 256 f32 scores (padded)
    _Float16* sA    = (_Float16*)(smem + OFF_A);       // 64 x 512 f16 tokens
    float*    sO    = (float*)smem;                    // 512 x 64 staging (reuse)
    float*    sMprev= (float*)(smem + OFF_STAT);       // [64] running max
    float*    sSum  = sMprev + 64;                     // [64] running denom
    float*    sG    = sMprev + 128;                    // [64] current base
    float*    sF    = sMprev + 192;                    // [64] rescale factor
    float*    sPart = (float*)(smem + OFF_PART);       // [256] reduction scratch

    const int tid  = threadIdx.x;
    const int w    = tid >> 5;          // wave 0..7
    const int lane = tid & 31;
    const int ll   = lane & 15;
    const int lh   = lane >> 4;

    const int bb    = blockIdx.x >> 6;          // batch
    const int nbase = (blockIdx.x & 63) * RROWS;

    // ---- load 64-row token tile as f16 ----
    for (int idx = tid; idx < RROWS * CDIM; idx += 256) {
        const int r = idx & 63, c = idx >> 6;
        sA[r * SA_STRIDE + c] =
            (_Float16)F[((size_t)(bb * CDIM + c)) * NTOK + nbase + r];
    }
    if (tid < 64) { sMprev[tid] = -1e30f; sSum[tid] = 0.0f; }

    v8f acc[4][4] = {};     // [row-tile][col-tile]: 64 rows x 64 cols of O
    __syncthreads();

    for (int sw = 0; sw < NSWEEP; ++sw) {
        // ===== pass 1: scores S = T * memory^T (f16 WMMA, f32 accum) =====
        // Wave owns 2 col-tiles; one B fetch feeds 4 row-tile WMMAs.
        for (int jt = 0; jt < 2; ++jt) {
            const int m0l = w * 32 + jt * 16;            // col base in sweep
            const int tm  = sw * 16 + w * 2 + jt;        // global m-tile
            v8f s[4] = {};
            #pragma unroll 2
            for (int k = 0; k < CDIM; k += 32) {
                const v16h b = ld_sw32(MhS + (((tm << 4) + (k >> 5)) << 9) + lane * 16);
                #pragma unroll
                for (int rt = 0; rt < 4; ++rt) {
                    const v16h a = ld_row32(sA + (rt * 16 + ll) * SA_STRIDE + k + lh * 8);
                    s[rt] = __builtin_amdgcn_wmma_f32_16x16x32_f16(false, a, false, b,
                                                                   (short)0, s[rt], false, false);
                }
            }
            #pragma unroll
            for (int rt = 0; rt < 4; ++rt)
                #pragma unroll
                for (int v = 0; v < 8; ++v)
                    sS[(rt * 16 + v + lh * 8) * SS_STRIDE + m0l + ll] = s[rt][v];
        }
        __syncthreads();

        // ===== pass 2a: per-row max of this sweep =====
        {
            const int r = tid >> 2, t = tid & 3;         // 4 threads per row
            const float* row = sS + r * SS_STRIDE;
            float mx = -1e30f;
            #pragma unroll 8
            for (int i = 0; i < 64; ++i) mx = fmaxf(mx, row[t + (i << 2)]);
            sPart[tid] = mx;
        }
        __syncthreads();
        if (tid < 64) {
            float mx = -1e30f;
            for (int i = 0; i < 4; ++i) mx = fmaxf(mx, sPart[tid * 4 + i]);
            const float g = fmaxf(sMprev[tid], mx);
            const float f = __expf(sMprev[tid] - g);     // <= 1, 0 on first sweep
            sG[tid] = g; sF[tid] = f;
            sMprev[tid] = g;
            sSum[tid] *= f;
        }
        __syncthreads();

        // flash-style rescale of the running output accumulators
        #pragma unroll
        for (int rt = 0; rt < 4; ++rt)
            #pragma unroll
            for (int v = 0; v < 8; ++v) {
                const float fv = sF[rt * 16 + lh * 8 + v];
                #pragma unroll
                for (int t = 0; t < 4; ++t) acc[rt][t][v] *= fv;
            }

        // ===== pass 2b: exp in place + per-row sums =====
        {
            const int r = tid >> 2, t = tid & 3;
            const float g = sG[r];
            float* row = sS + r * SS_STRIDE;
            float sm = 0.0f;
            #pragma unroll 8
            for (int i = 0; i < 64; ++i) {
                const float e = __expf(row[t + (i << 2)] - g);
                row[t + (i << 2)] = e;
                sm += e;
            }
            sPart[tid] = sm;
        }
        __syncthreads();
        if (tid < 64) {
            float sm = 0.0f;
            for (int i = 0; i < 4; ++i) sm += sPart[tid * 4 + i];
            sSum[tid] += sm;
        }
        __syncthreads();

        // ===== pass 3: O += P * memory (f16 P on the fly, f32 accum) =====
        for (int kc = 0; kc < SWEEP; kc += 32) {
            v16h a[4];
            #pragma unroll
            for (int rt = 0; rt < 4; ++rt) {
                const float* ap = sS + (rt * 16 + ll) * SS_STRIDE + kc + lh * 8;
                const float4 q0 = *(const float4*)(ap);
                const float4 q1 = *(const float4*)(ap + 4);
                const float4 q2 = *(const float4*)(ap + 16);
                const float4 q3 = *(const float4*)(ap + 20);
                const float tmp[16] = { q0.x, q0.y, q0.z, q0.w, q1.x, q1.y, q1.z, q1.w,
                                        q2.x, q2.y, q2.z, q2.w, q3.x, q3.y, q3.z, q3.w };
                #pragma unroll
                for (int i = 0; i < 16; ++i) a[rt][i] = (_Float16)tmp[i];
            }
            const int tk2 = sw * 8 + (kc >> 5);
            #pragma unroll
            for (int t = 0; t < 4; ++t) {
                const int tc = (w << 2) + t;
                const v16h bt = ld_sw32(MtS + (((tc << 6) + tk2) << 9) + lane * 16);
                #pragma unroll
                for (int rt = 0; rt < 4; ++rt)
                    acc[rt][t] = __builtin_amdgcn_wmma_f32_16x16x32_f16(false, a[rt], false, bt,
                                                                        (short)0, acc[rt][t], false, false);
            }
        }
        __syncthreads();   // before next sweep overwrites sS
    }

    // ---- normalize (fold 1/sum) and stage transposed for coalesced stores ----
    #pragma unroll
    for (int rt = 0; rt < 4; ++rt)
        #pragma unroll
        for (int v = 0; v < 8; ++v) {
            const float inv = 1.0f / sSum[rt * 16 + lh * 8 + v];
            #pragma unroll
            for (int t = 0; t < 4; ++t) {
                const int c = w * 64 + t * 16 + ll;
                sO[c * 65 + rt * 16 + lh * 8 + v] = acc[rt][t][v] * inv;
            }
        }
    __syncthreads();
    for (int idx = tid; idx < RROWS * CDIM; idx += 256) {
        const int r = idx & 63, c = idx >> 6;
        Out[((size_t)(bb * CDIM + c)) * NTOK + nbase + r] = sO[c * 65 + r];
    }
}

extern "C" void kernel_launch(void* const* d_in, const int* in_sizes, int n_in,
                              void* d_out, int out_size, void* d_ws, size_t ws_size,
                              hipStream_t stream) {
    const float* F   = (const float*)d_in[0];   // (16, 512, 64, 64) f32
    const float* Mem = (const float*)d_in[1];   // (2048, 512) f32
    float* Out = (float*)d_out;

    _Float16* mhW = (_Float16*)d_ws;                 // 2 MB swizzled bank (K=C)
    _Float16* mtW = mhW + (size_t)MDIM * CDIM;       // 2 MB swizzled bank (K=M)

    mem_swizzle_f16_kernel<<<(MDIM * CDIM) / 256, 256, 0, stream>>>(Mem, mhW, mtW);
    mem_attn_kernel<<<16 * (NTOK / RROWS), 256, 0, stream>>>(F, mhW, mtW, Out);
}
